// SupConLoss_74354473828394
// MI455X (gfx1250) — matrix operand order repaired and checked by reading, not compiled, not run on hardware
//
#include <hip/hip_runtime.h>
#include <hip/hip_bf16.h>

typedef _Float16 half_t;
typedef __attribute__((ext_vector_type(16))) _Float16 v16h;
typedef __attribute__((ext_vector_type(8)))  _Float16 v8h;
typedef __attribute__((ext_vector_type(8)))  float    v8f;

#define N_TOT   12544      // B*H*W
#define D_FEAT  256
#define D_Z     128
#define S_ANC   196        // SPLIT / TRAINPATCHSPLIT
#define S_PAD   208        // 13 * 16 (anchor rows padded to tile multiple)
#define SPLITC  784
#define MI_CNT  16
#define PROP    3
#define NT      (N_TOT / 16)   // 784 n-tiles
#define MT      (S_PAD / 16)   // 13 m-tiles
#define POS0    0.1f
#define AMB0    0.05f
#define TEMP_C  0.07f

// ---- WMMA fragment loads (16x16x32 f16, contiguous k order shared by A and B) ----
// A (16x32, row-major source): lane holds row M = lane&15.
//   lane<16 : halves 0..15 = K0..15
//   lane>=16: halves 0..15 = K16..31
__device__ inline v16h load_a16(const half_t* A, size_t lda, int row, int k0, int hi) {
    return *(const v16h*)(A + (size_t)row * lda + (size_t)(k0 + hi * 16));
}
// B (32x16) where B[k,j] = X[n0+j, k] (X row-major along K):
//   lane<16 : column = lane, halves = K0..15  -> one contiguous 32B load
//   lane>=16: column = lane-16, halves = K16..31
__device__ inline v16h load_b16(const half_t* X, size_t ldx, int n0, int k0, int lr, int hi) {
    return *(const v16h*)(X + (size_t)(n0 + lr) * ldx + (size_t)(k0 + hi * 16));
}

__device__ inline float blockReduce256(float x, float* sh) {
    int t = threadIdx.x;
    sh[t] = x; __syncthreads();
    for (int off = 128; off > 0; off >>= 1) {
        if (t < off) sh[t] += sh[t + off];
        __syncthreads();
    }
    float r = sh[0]; __syncthreads();
    return r;
}

// ---- Prep: normalize feat (B,D,H,W)->(N,D) f16 row-major + transposed ----
__global__ void __launch_bounds__(256)
k_prep_feat(const float* __restrict__ feat, half_t* __restrict__ feat16,
            half_t* __restrict__ featT16) {
    int n = blockIdx.x, d = threadIdx.x;
    int b = n / SPLITC, r = n % SPLITC, h = r / 28, w = r % 28;
    float v = feat[(((size_t)b * D_FEAT + d) * 28 + h) * 28 + w];
    __shared__ float sh[256];
    float ss = blockReduce256(v * v, sh);
    float inv = 1.0f / fmaxf(sqrtf(ss), 1e-12f);
    float x = v * inv;
    feat16[(size_t)n * D_FEAT + d]  = (half_t)x;
    featT16[(size_t)d * N_TOT + n]  = (half_t)x;
}

__global__ void k_prep_z(const float* __restrict__ z, half_t* __restrict__ z16, int total) {
    int i = blockIdx.x * blockDim.x + threadIdx.x;
    if (i < total) z16[i] = (half_t)z[i];
}

// ---- Per-mi init: Psi/Phi constants, proxy_prev = gathered feat rows ----
__global__ void __launch_bounds__(256)
k_init(const half_t* __restrict__ feat16, const int* __restrict__ sidx, int mi,
       float* __restrict__ proxyPrev, float* __restrict__ Psi, float* __restrict__ Phi,
       float* __restrict__ accum) {
    int s = blockIdx.x;                       // 0..207
    int sc = (s < S_ANC) ? s : (S_ANC - 1);
    int row = mi * SPLITC + sidx[mi * S_ANC + sc];
    proxyPrev[(size_t)s * D_FEAT + threadIdx.x] =
        (float)feat16[(size_t)row * D_FEAT + threadIdx.x];
    if (threadIdx.x == 0) { Psi[s] = AMB0; Phi[s] = POS0; }
    if (s == 0 && threadIdx.x < 4) accum[mi * 4 + threadIdx.x] = 0.0f;
}

// ---- cos0 = feat[rows] @ feat.T ; write out_cos f32 and mask16 = cos0 > POS0 ----
__global__ void __launch_bounds__(128)
k_cos0(const half_t* __restrict__ feat16, const int* __restrict__ sidx, int mi,
       float* __restrict__ outcos, half_t* __restrict__ mask16) {
    const int wave = threadIdx.x >> 5, lane = threadIdx.x & 31;
    const int lr = lane & 15, hi = lane >> 4;
    const int nt = blockIdx.x * 4 + wave;   // 0..783
    const int m0 = blockIdx.y * 16;
    int m = m0 + lr; if (m >= S_ANC) m = S_ANC - 1;
    const int arow = mi * SPLITC + sidx[mi * S_ANC + m];
    v8f acc = {};
    for (int k0 = 0; k0 < D_FEAT; k0 += 32) {
        v16h a = load_a16(feat16, D_FEAT, arow, k0, hi);
        v16h b = load_b16(feat16, D_FEAT, nt * 16, k0, lr, hi);
        acc = __builtin_amdgcn_wmma_f32_16x16x32_f16(false, a, false, b, (short)0, acc, false, false);
    }
    const int n = nt * 16 + lr;
#pragma unroll
    for (int v = 0; v < 8; v++) {
        int mr = m0 + v + 8 * hi;
        float oc = acc[v];
        outcos[(size_t)mr * N_TOT + n] = oc;
        mask16[(size_t)mr * N_TOT + n] = (oc > POS0) ? (half_t)1.0f : (half_t)0.0f;
    }
}

// ---- proxy_raw = (mask16 @ feat) / N  (K = 12544, via transposed-feat B loads) ----
__global__ void __launch_bounds__(128)
k_proxy(const half_t* __restrict__ mask16, const half_t* __restrict__ featT16,
        float* __restrict__ proxy32) {
    const int wave = threadIdx.x >> 5, lane = threadIdx.x & 31;
    const int lr = lane & 15, hi = lane >> 4;
    const int dt = blockIdx.x * 4 + wave;   // 0..15
    const int m0 = blockIdx.y * 16;
    v8f acc = {};
    for (int k0 = 0; k0 < N_TOT; k0 += 32) {
        v16h a = load_a16(mask16, N_TOT, m0 + lr, k0, hi);
        v16h b = load_b16(featT16, N_TOT, dt * 16, k0, lr, hi);
        __builtin_prefetch(featT16 + (size_t)(dt * 16 + lr) * N_TOT + k0 + 256, 0, 1);
        acc = __builtin_amdgcn_wmma_f32_16x16x32_f16(false, a, false, b, (short)0, acc, false, false);
    }
    const int d = dt * 16 + lr;
#pragma unroll
    for (int v = 0; v < 8; v++)
        proxy32[(size_t)(m0 + v + 8 * hi) * D_FEAT + d] = acc[v] * (1.0f / (float)N_TOT);
}

// ---- normalize proxy, cossim vs prev, update Psi/Phi, emit proxy16/prev ----
__global__ void __launch_bounds__(256)
k_norm(const float* __restrict__ proxy32, float* __restrict__ proxyPrev,
       half_t* __restrict__ proxy16, float* __restrict__ Psi, float* __restrict__ Phi) {
    int s = blockIdx.x, d = threadIdx.x;
    float v  = proxy32[(size_t)s * D_FEAT + d];
    float pv = proxyPrev[(size_t)s * D_FEAT + d];
    __shared__ float sh[256];
    float ss = blockReduce256(v * v, sh);
    float dd = blockReduce256(v * pv, sh);
    float pp = blockReduce256(pv * pv, sh);
    float nrm = sqrtf(ss), np = sqrtf(pp);
    float ms = dd / fmaxf(nrm * np, 1e-8f);
    if (d == 0) { float delta = 1.0f - ms; Psi[s] += delta; Phi[s] -= delta; }
    float inv = 1.0f / fmaxf(nrm, 1e-12f);
    float x = v * inv;
    proxy16[(size_t)s * D_FEAT + d]  = (half_t)x;
    proxyPrev[(size_t)s * D_FEAT + d] = x;
}

// ---- out_cos = proxy @ feat.T ; mask16 = out_cos > Phi[row] ----
__global__ void __launch_bounds__(128)
k_outcos(const half_t* __restrict__ proxy16, const float* __restrict__ Phi,
         const half_t* __restrict__ feat16, float* __restrict__ outcos,
         half_t* __restrict__ mask16) {
    const int wave = threadIdx.x >> 5, lane = threadIdx.x & 31;
    const int lr = lane & 15, hi = lane >> 4;
    const int nt = blockIdx.x * 4 + wave;
    const int m0 = blockIdx.y * 16;
    v8f acc = {};
    for (int k0 = 0; k0 < D_FEAT; k0 += 32) {
        v16h a = load_a16(proxy16, D_FEAT, m0 + lr, k0, hi);
        v16h b = load_b16(feat16, D_FEAT, nt * 16, k0, lr, hi);
        acc = __builtin_amdgcn_wmma_f32_16x16x32_f16(false, a, false, b, (short)0, acc, false, false);
    }
    const int n = nt * 16 + lr;
#pragma unroll
    for (int v = 0; v < 8; v++) {
        int mr = m0 + v + 8 * hi;
        float oc = acc[v];
        outcos[(size_t)mr * N_TOT + n] = oc;
        mask16[(size_t)mr * N_TOT + n] = (oc > Phi[mr]) ? (half_t)1.0f : (half_t)0.0f;
    }
}

// ---- logits = z[rows] @ z.T / TEMP  (K = 128) ----
__global__ void __launch_bounds__(128)
k_logits(const half_t* __restrict__ z16, const int* __restrict__ sidx, int mi,
         float* __restrict__ logits32) {
    const int wave = threadIdx.x >> 5, lane = threadIdx.x & 31;
    const int lr = lane & 15, hi = lane >> 4;
    const int nt = blockIdx.x * 4 + wave;
    const int m0 = blockIdx.y * 16;
    int m = m0 + lr; if (m >= S_ANC) m = S_ANC - 1;
    const int arow = mi * SPLITC + sidx[mi * S_ANC + m];
    v8f acc = {};
#pragma unroll
    for (int k0 = 0; k0 < D_Z; k0 += 32) {
        v16h a = load_a16(z16, D_Z, arow, k0, hi);
        v16h b = load_b16(z16, D_Z, nt * 16, k0, lr, hi);
        acc = __builtin_amdgcn_wmma_f32_16x16x32_f16(false, a, false, b, (short)0, acc, false, false);
    }
    const int n = nt * 16 + lr;
#pragma unroll
    for (int v = 0; v < 8; v++)
        logits32[(size_t)(m0 + v + 8 * hi) * N_TOT + n] = acc[v] * (1.0f / TEMP_C);
}

// ---- per-anchor masked log-softmax loss, accumulate per-mi stats ----
__global__ void __launch_bounds__(256)
k_loss(const float* __restrict__ outcos, const float* __restrict__ logits,
       const float* __restrict__ Psi, const float* __restrict__ Phi,
       const int* __restrict__ sidx, int mi, float* __restrict__ accum) {
    int s = blockIdx.x;                // 0..195
    int t = threadIdx.x;
    int self = mi * SPLITC + sidx[mi * S_ANC + s];
    float psi = Psi[s], phi = Phi[s];
    const float* Lrow = logits + (size_t)s * N_TOT;
    const float* Crow = outcos + (size_t)s * N_TOT;
    __shared__ float sh[256];
    // pass 1: full-row max of logits
    float lm = -1e30f;
    for (int n = t; n < N_TOT; n += 256) lm = fmaxf(lm, Lrow[n]);
    sh[t] = lm; __syncthreads();
    for (int off = 128; off > 0; off >>= 1) {
        if (t < off) sh[t] = fmaxf(sh[t], sh[t + off]);
        __syncthreads();
    }
    float M = sh[0]; __syncthreads();
    // pass 2: masked sums
    float dsum = 0.0f, sml = 0.0f, msum = 0.0f, neg = 0.0f;
    for (int n = t; n < N_TOT; n += 256) {
        float l = Lrow[n], oc = Crow[n];
        float mnew = (oc > phi) ? 1.0f : 0.0f;
        float nb   = (psi > oc) ? 1.0f : 0.0f;
        float lmk  = (n != self) ? 1.0f : 0.0f;
        float mk   = mnew * lmk;
        float ng   = ((mk + nb) > 0.0f) ? lmk : 0.0f;
        dsum += expf(l - M) * ng;
        sml  += mk * l;
        msum += mk;
        neg  += nb;
    }
    dsum = blockReduce256(dsum, sh);
    sml  = blockReduce256(sml, sh);
    msum = blockReduce256(msum, sh);
    neg  = blockReduce256(neg, sh);
    if (t == 0) {
        float valid = (msum > 0.0f) ? 1.0f : 0.0f;
        float mlp = 0.0f;
        if (msum > 0.0f)
            mlp = (sml - msum * (M + logf(fmaxf(dsum, 1e-38f)))) / msum;
        atomicAdd(&accum[mi * 4 + 0], mlp * valid);
        atomicAdd(&accum[mi * 4 + 1], valid);
        atomicAdd(&accum[mi * 4 + 2], msum);
        atomicAdd(&accum[mi * 4 + 3], neg);
    }
}

__global__ void k_final(const float* __restrict__ accum, float* __restrict__ out) {
    if (blockIdx.x == 0 && threadIdx.x == 0) {
        float L = 0.0f, P = 0.0f, G = 0.0f;
        for (int mi = 0; mi < MI_CNT; mi++) {
            float sm = accum[mi * 4 + 0], nv = accum[mi * 4 + 1];
            float mk = accum[mi * 4 + 2], ng = accum[mi * 4 + 3];
            L += -(TEMP_C / 0.07f) * sm / fmaxf(nv, 1.0f);
            P += mk / (float)S_ANC;
            G += ng / (float)S_ANC;
        }
        out[0] = L / (float)MI_CNT;
        out[1] = P / (float)MI_CNT;
        out[2] = G / (float)MI_CNT;
    }
}

extern "C" void kernel_launch(void* const* d_in, const int* in_sizes, int n_in,
                              void* d_out, int out_size, void* d_ws, size_t ws_size,
                              hipStream_t stream) {
    (void)in_sizes; (void)n_in; (void)out_size; (void)ws_size;
    const float* z    = (const float*)d_in[0];
    const float* feat = (const float*)d_in[1];
    const int*   sidx = (const int*)d_in[2];
    float* out = (float*)d_out;

    char* ws = (char*)d_ws;
    size_t off = 0;
    auto alloc = [&](size_t bytes) -> char* {
        char* p = ws + off;
        off = (off + bytes + 255) & ~(size_t)255;
        return p;
    };
    half_t* feat16    = (half_t*)alloc((size_t)N_TOT * D_FEAT * sizeof(half_t));
    half_t* featT16   = (half_t*)alloc((size_t)D_FEAT * N_TOT * sizeof(half_t));
    half_t* z16       = (half_t*)alloc((size_t)N_TOT * D_Z * sizeof(half_t));
    half_t* mask16    = (half_t*)alloc((size_t)S_PAD * N_TOT * sizeof(half_t));
    float*  outcos    = (float*)alloc((size_t)S_PAD * N_TOT * sizeof(float));
    float*  logits32  = (float*)alloc((size_t)S_PAD * N_TOT * sizeof(float));
    float*  proxy32   = (float*)alloc((size_t)S_PAD * D_FEAT * sizeof(float));
    float*  proxyPrev = (float*)alloc((size_t)S_PAD * D_FEAT * sizeof(float));
    half_t* proxy16   = (half_t*)alloc((size_t)S_PAD * D_FEAT * sizeof(half_t));
    float*  Psi       = (float*)alloc((size_t)S_PAD * sizeof(float));
    float*  Phi       = (float*)alloc((size_t)S_PAD * sizeof(float));
    float*  accum     = (float*)alloc((size_t)MI_CNT * 4 * sizeof(float));

    k_prep_feat<<<N_TOT, 256, 0, stream>>>(feat, feat16, featT16);
    k_prep_z<<<(N_TOT * D_Z + 255) / 256, 256, 0, stream>>>(z, z16, N_TOT * D_Z);

    for (int mi = 0; mi < MI_CNT; ++mi) {
        k_init<<<S_PAD, 256, 0, stream>>>(feat16, sidx, mi, proxyPrev, Psi, Phi, accum);
        k_cos0<<<dim3(NT / 4, MT), 128, 0, stream>>>(feat16, sidx, mi, outcos, mask16);
        for (int it = 0; it < PROP; ++it) {
            k_proxy<<<dim3(4, MT), 128, 0, stream>>>(mask16, featT16, proxy32);
            k_norm<<<S_PAD, 256, 0, stream>>>(proxy32, proxyPrev, proxy16, Psi, Phi);
            k_outcos<<<dim3(NT / 4, MT), 128, 0, stream>>>(proxy16, Phi, feat16, outcos, mask16);
        }
        k_logits<<<dim3(NT / 4, MT), 128, 0, stream>>>(z16, sidx, mi, logits32);
        k_loss<<<S_ANC, 256, 0, stream>>>(outcos, logits32, Psi, Phi, sidx, mi, accum);
    }
    k_final<<<1, 32, 0, stream>>>(accum, out);
}
